// SoftModularActorCritic_88682484727918
// MI455X (gfx1250) — matrix operand order repaired and checked by reading, not compiled, not run on hardware
//
#include <hip/hip_runtime.h>
#include <hip/hip_bf16.h>

// Problem sizes
#define B_    16384
#define OBS_  376
#define OBSP_ 384      // OBS padded to multiple of 32 (WMMA K step)
#define H_    512
#define M_    8
#define L1_   3
#define A_    17
#define OUTC_ (A_ + 3) // mean(17) | log_prob | entropy | value

typedef __bf16 v16bf __attribute__((ext_vector_type(16)));
typedef float  v8f   __attribute__((ext_vector_type(8)));

// ---- CDNA5 async global->LDS copy (ASYNCcnt-tracked), with safe fallback ----
#if defined(__has_builtin)
# if __has_builtin(__builtin_amdgcn_global_load_async_to_lds_b128)
#  define ASYNC_OK 1
# endif
#endif
#ifndef ASYNC_OK
# define ASYNC_OK 0
#endif

typedef int async_v4i __attribute__((vector_size(16)));
typedef __attribute__((address_space(1))) async_v4i* g_v4i_ptr;
typedef __attribute__((address_space(3))) async_v4i* l_v4i_ptr;

__device__ __forceinline__ void async_cp16(const unsigned short* g, unsigned short* l) {
#if ASYNC_OK
  __builtin_amdgcn_global_load_async_to_lds_b128((g_v4i_ptr)g, (l_v4i_ptr)l, 0, 0);
#else
  *(uint4*)l = *(const uint4*)g;
#endif
}
__device__ __forceinline__ void wait_async() {
#if ASYNC_OK
# if __has_builtin(__builtin_amdgcn_s_wait_asynccnt)
  __builtin_amdgcn_s_wait_asynccnt(0);
# else
  asm volatile("s_wait_asynccnt 0x0" ::: "memory");
# endif
#endif
}

__device__ __forceinline__ unsigned short f2bf(float f) {
  union { float f; unsigned int u; } v; v.f = f;
  unsigned int r = v.u + 0x7FFFu + ((v.u >> 16) & 1u);  // round-to-nearest-even
  return (unsigned short)(r >> 16);
}
__device__ __forceinline__ float bf2f(unsigned short s) {
  union { float f; unsigned int u; } v; v.u = ((unsigned int)s) << 16;
  return v.f;
}

// ---------------- conversion kernels ----------------
__global__ void cvt_pad_cols(const float* __restrict__ s, unsigned short* __restrict__ d,
                             int rows, int cols, int colsP) {
  long i = (long)blockIdx.x * blockDim.x + threadIdx.x;
  if (i >= (long)rows * colsP) return;
  int r = (int)(i / colsP), c = (int)(i % colsP);
  d[i] = f2bf(c < cols ? s[(long)r * cols + c] : 0.0f);
}
// batched weight transpose: src[b][K][N] (f32) -> dst[b][N][Kp] (bf16, K padded)
__global__ void cvt_transpose_batched(const float* __restrict__ s, unsigned short* __restrict__ d,
                                      int batch, int K, int N, int Kp) {
  long i = (long)blockIdx.x * blockDim.x + threadIdx.x;
  long per = (long)N * Kp;
  if (i >= (long)batch * per) return;
  int b = (int)(i / per);
  long r = i % per;
  int n = (int)(r / Kp), k = (int)(r % Kp);
  d[i] = f2bf(k < K ? s[(long)b * K * N + (long)k * N + n] : 0.0f);
}

// ---------------- generic bf16 WMMA GEMM (double-buffered async pipeline) ----
// C(gm,gn) = act( sum_k A[gm,k]*Wt[gn,k] + bias[gn] )   [Wt pre-transposed [N][K]]
// block tile: 128 (M) x 64 (N); 8 waves, each wave -> 32x32 (2x2 WMMA 16x16x32)
// blockIdx.z = module index (per-module strides sAm/sWm/sBm/sOm)
__launch_bounds__(256)
__global__ void gemm_bf16_wmma(const unsigned short* __restrict__ A, int lda, long sAm,
                               const unsigned short* __restrict__ Wt, int ldw, long sWm,
                               const float* __restrict__ bias, int sBm,
                               unsigned short* __restrict__ outB, float* __restrict__ outF,
                               int ldo, long sOm, int K, int act) {
  __shared__ __align__(16) unsigned short sA[2][128 * 32];  // [row][k]
  __shared__ __align__(16) unsigned short sB[2][64 * 32];   // [col][k] (K-contiguous)

  const int t    = threadIdx.x;
  const int lane = t & 31;
  const int wave = t >> 5;
  const int m    = blockIdx.z;

  const unsigned short* Ab = A  + (long)blockIdx.x * 128 * lda + (long)m * sAm;
  const unsigned short* Wb = Wt + (long)blockIdx.y * 64 * ldw  + (long)m * sWm;

  const int mOff = (wave >> 1) * 32;
  const int nOff = (wave & 1) * 32;

  v8f acc[2][2];
#pragma unroll
  for (int i = 0; i < 2; i++)
#pragma unroll
    for (int j = 0; j < 2; j++)
#pragma unroll
      for (int r = 0; r < 8; r++) acc[i][j][r] = 0.0f;

  // staging assignments: A tile 128x32 (2x16B/thread), B tile 64x32 (1x16B/thread)
  const int aRow  = t >> 1;        // 0..127
  const int aHalf = (t & 1) * 16;  // 0 or 16
  const int bN    = t >> 2;        // 0..63
  const int bSeg  = (t & 3) * 8;   // 0,8,16,24

  const int nK = K >> 5;

  // prologue: stage tile 0 into buffer 0
  {
    const unsigned short* ga = Ab + (long)aRow * lda + aHalf;
    async_cp16(ga,     &sA[0][aRow * 32 + aHalf]);
    async_cp16(ga + 8, &sA[0][aRow * 32 + aHalf + 8]);
    async_cp16(Wb + (long)bN * ldw + bSeg, &sB[0][bN * 32 + bSeg]);
  }
  wait_async();
  __syncthreads();

  for (int kt = 0; kt < nK; kt++) {
    const int cur = kt & 1;
    if (kt + 1 < nK) {  // issue next tile's async copies into the other buffer
      const int nxt = cur ^ 1;
      const int k0 = (kt + 1) * 32;
      const unsigned short* ga = Ab + (long)aRow * lda + k0 + aHalf;
      async_cp16(ga,     &sA[nxt][aRow * 32 + aHalf]);
      async_cp16(ga + 8, &sA[nxt][aRow * 32 + aHalf + 8]);
      async_cp16(Wb + (long)bN * ldw + k0 + bSeg, &sB[nxt][bN * 32 + bSeg]);
    }

    // fragment loads: A and B both in contiguous k order (k = 16*(lane/16) + element)
    const int a_b0  = (lane < 16) ? 0 : 16;
    const int rowA0 = mOff + (lane & 15);
    const int colB0 = nOff + (lane & 15);
    const int kbB   = (lane >> 4) * 16;

    union { v16bf v; uint4 q[2]; } fa[2], fb[2];
#pragma unroll
    for (int i = 0; i < 2; i++) {
      const unsigned short* ap = &sA[cur][(rowA0 + i * 16) * 32];
      fa[i].q[0] = *(const uint4*)(ap + a_b0);
      fa[i].q[1] = *(const uint4*)(ap + a_b0 + 8);
    }
#pragma unroll
    for (int j = 0; j < 2; j++) {
      const unsigned short* bp = &sB[cur][(colB0 + j * 16) * 32 + kbB];
      fb[j].q[0] = *(const uint4*)(bp);
      fb[j].q[1] = *(const uint4*)(bp + 8);
    }
#pragma unroll
    for (int i = 0; i < 2; i++)
#pragma unroll
      for (int j = 0; j < 2; j++)
        acc[i][j] = __builtin_amdgcn_wmma_f32_16x16x32_bf16(
            false, fa[i].v, false, fb[j].v, (short)0, acc[i][j], false, false);

    wait_async();     // next buffer fully staged (this wave's async ops done)
    __syncthreads();  // everyone done reading `cur` before it is overwritten
  }

  // epilogue: C/D layout: VGPR r -> row = r + 8*(lane>=16), col = lane%16
  const int rBase = (lane >> 4) * 8;
  const int cLane = lane & 15;
#pragma unroll
  for (int i = 0; i < 2; i++) {
#pragma unroll
    for (int j = 0; j < 2; j++) {
      const int gn = blockIdx.y * 64 + nOff + j * 16 + cLane;
      float bv = bias ? bias[(long)m * sBm + gn] : 0.0f;
#pragma unroll
      for (int r = 0; r < 8; r++) {
        float v = acc[i][j][r] + bv;
        if (act == 1)      v = fmaxf(v, 0.0f);
        else if (act == 2) v = tanhf(v);
        const long gm = (long)blockIdx.x * 128 + mOff + i * 16 + rBase + r;
        if (outB) outB[gm * ldo + (long)m * sOm + gn] = f2bf(v);
        if (outF) outF[gm * ldo + (long)m * sOm + gn] = v;
      }
    }
  }
}

// ---------------- small kernels ----------------
__global__ void context_relu(const unsigned short* __restrict__ fs, const int* __restrict__ tidx,
                             const float* __restrict__ temb, unsigned short* __restrict__ relc) {
  long i = (long)blockIdx.x * blockDim.x + threadIdx.x;
  if (i >= (long)B_ * H_) return;
  int b = (int)(i >> 9), h = (int)(i & 511);
  float v = bf2f(fs[i]) * temb[(long)tidx[b] * H_ + h];
  relc[i] = f2bf(fmaxf(v, 0.0f));
}

__global__ void softmax8_inplace(float* __restrict__ p, int ngroups) {
  int i = blockIdx.x * blockDim.x + threadIdx.x;
  if (i >= ngroups) return;
  float* q = p + (long)i * 8;
  float mx = q[0];
#pragma unroll
  for (int j = 1; j < 8; j++) mx = fmaxf(mx, q[j]);
  float e[8], s = 0.0f;
#pragma unroll
  for (int j = 0; j < 8; j++) { e[j] = __expf(q[j] - mx); s += e[j]; }
  float inv = 1.0f / s;
#pragma unroll
  for (int j = 0; j < 8; j++) q[j] = e[j] * inv;
}

// g[b,i,h] = sum_j probs[b,i,j] * outs[b,j,h]
__launch_bounds__(256)
__global__ void combine_kernel(const float* __restrict__ probs, const unsigned short* __restrict__ outs,
                               unsigned short* __restrict__ g) {
  __shared__ __align__(16) unsigned short sO[M_ * H_];
  __shared__ float sP[64];
  const int b = blockIdx.x, t = threadIdx.x;
  {
    const uint4* src = (const uint4*)(outs + (long)b * (M_ * H_)) + t * 2;
    uint4 v0 = src[0], v1 = src[1];
    uint4* dst = (uint4*)sO + t * 2;
    dst[0] = v0; dst[1] = v1;
  }
  if (t < 64) sP[t] = probs[(long)b * 64 + t];
  __syncthreads();
  unsigned short* gb = g + (long)b * (M_ * H_);
  for (int idx = t; idx < M_ * H_; idx += 256) {
    int i = idx >> 9, h = idx & 511;
    float s = 0.0f;
#pragma unroll
    for (int j = 0; j < 8; j++) s += sP[i * 8 + j] * bf2f(sO[j * H_ + h]);
    gb[idx] = f2bf(s);
  }
}

__launch_bounds__(256)
__global__ void actor_final(const unsigned short* __restrict__ g, const float* __restrict__ Wf,
                            const float* __restrict__ bf, const float* __restrict__ logstd,
                            float* __restrict__ out) {
  __shared__ __align__(16) unsigned short sG[M_ * H_];
  const int b = blockIdx.x, t = threadIdx.x, lane = t & 31, wave = t >> 5;
  {
    const uint4* src = (const uint4*)(g + (long)b * (M_ * H_)) + t * 2;
    uint4 v0 = src[0], v1 = src[1];
    uint4* dst = (uint4*)sG + t * 2;
    dst[0] = v0; dst[1] = v1;
  }
  __syncthreads();
  float* ob = out + (long)b * OUTC_;
  for (int a = wave; a < A_; a += 8) {
    float s = 0.0f;
    for (int f = lane; f < M_ * H_; f += 32) {
      int mm = f >> 9, h = f & 511;
      s += bf2f(sG[f]) * Wf[((long)mm * H_ + h) * A_ + a];
    }
#pragma unroll
    for (int off = 16; off > 0; off >>= 1) s += __shfl_down(s, off, 32);
    if (lane == 0) {
      float bs = 0.0f;
#pragma unroll
      for (int mm = 0; mm < M_; mm++) bs += bf[mm * A_ + a];
      ob[a] = s + bs;
    }
  }
  if (t == 0) {
    float sl = 0.0f;
#pragma unroll
    for (int a = 0; a < A_; a++) sl += logstd[a];
    const float C = 0.91893853320467274f;  // 0.5*ln(2*pi)
    ob[A_]     = -sl - A_ * C;             // log_prob
    ob[A_ + 1] =  sl + A_ * (0.5f + C);    // entropy
  }
}

__launch_bounds__(256)
__global__ void critic_final(const unsigned short* __restrict__ g, const float* __restrict__ Wf,
                             const float* __restrict__ bf, float* __restrict__ out) {
  __shared__ __align__(16) unsigned short sG[M_ * H_];
  __shared__ float red[256];
  const int b = blockIdx.x, t = threadIdx.x;
  {
    const uint4* src = (const uint4*)(g + (long)b * (M_ * H_)) + t * 2;
    uint4 v0 = src[0], v1 = src[1];
    uint4* dst = (uint4*)sG + t * 2;
    dst[0] = v0; dst[1] = v1;
  }
  __syncthreads();
  float s = 0.0f;
  for (int f = t; f < M_ * H_; f += 256) {
    int mm = f >> 9, h = f & 511;
    s += bf2f(sG[f]) * Wf[(long)mm * H_ + h];
  }
  red[t] = s; __syncthreads();
  for (int off = 128; off > 0; off >>= 1) {
    if (t < off) red[t] += red[t + off];
    __syncthreads();
  }
  if (t == 0) {
    float bs = 0.0f;
#pragma unroll
    for (int mm = 0; mm < M_; mm++) bs += bf[mm];
    out[(long)b * OUTC_ + A_ + 2] = red[0] + bs;
  }
}

// ---------------- host launch ----------------
extern "C" void kernel_launch(void* const* d_in, const int* in_sizes, int n_in,
                              void* d_out, int out_size, void* d_ws, size_t ws_size,
                              hipStream_t stream) {
  (void)in_sizes; (void)n_in; (void)out_size; (void)ws_size;
  const float* x        = (const float*)d_in[0];
  const int*   task_idx = (const int*)d_in[1];
  const float* enc_W1   = (const float*)d_in[2];
  const float* enc_b1   = (const float*)d_in[3];
  const float* enc_W2   = (const float*)d_in[4];
  const float* enc_b2   = (const float*)d_in[5];
  const float* task_emb = (const float*)d_in[6];
  const float* a_rW     = (const float*)d_in[7];
  const float* a_rb     = (const float*)d_in[8];
  const float* c_rW     = (const float*)d_in[9];
  const float* c_rb     = (const float*)d_in[10];
  const float* a_W      = (const float*)d_in[11];
  const float* a_b      = (const float*)d_in[12];
  const float* a_Wf     = (const float*)d_in[13];
  const float* a_bf     = (const float*)d_in[14];
  const float* c_W      = (const float*)d_in[15];
  const float* c_b      = (const float*)d_in[16];
  const float* c_Wf     = (const float*)d_in[17];
  const float* c_bf     = (const float*)d_in[18];
  const float* logstd   = (const float*)d_in[19];
  float* out = (float*)d_out;

  size_t off = 0;
  char* base = (char*)d_ws;
  auto alloc = [&](size_t bytes) -> void* {
    void* p = base + off;
    off += (bytes + 255) & ~(size_t)255;
    return p;
  };
  unsigned short* xb     = (unsigned short*)alloc((size_t)B_ * OBSP_ * 2);
  unsigned short* encW1t = (unsigned short*)alloc((size_t)H_ * OBSP_ * 2);     // [512][384]
  unsigned short* encW2t = (unsigned short*)alloc((size_t)H_ * H_ * 2);        // [512][512]
  unsigned short* h1     = (unsigned short*)alloc((size_t)B_ * H_ * 2);
  unsigned short* fs     = (unsigned short*)alloc((size_t)B_ * H_ * 2);
  unsigned short* relc   = (unsigned short*)alloc((size_t)B_ * H_ * 2);
  unsigned short* a_rWt  = (unsigned short*)alloc((size_t)L1_ * 64 * H_ * 2);  // [3][64][512]
  unsigned short* c_rWt  = (unsigned short*)alloc((size_t)L1_ * 64 * H_ * 2);
  unsigned short* a_Wt   = (unsigned short*)alloc((size_t)L1_ * M_ * H_ * H_ * 2);
  unsigned short* c_Wt   = (unsigned short*)alloc((size_t)L1_ * M_ * H_ * H_ * 2);
  float*          probs  = (float*)alloc((size_t)L1_ * B_ * 64 * 4);
  unsigned short* G      = (unsigned short*)alloc((size_t)B_ * M_ * H_ * 2);
  unsigned short* O      = (unsigned short*)alloc((size_t)B_ * M_ * H_ * 2);

  const int TPB = 256;
  auto nb = [](long n) { return (unsigned)((n + 255) / 256); };

  // input / weight conversion to bf16 (weights transposed to [N][K] for async staging)
  cvt_pad_cols<<<nb((long)B_ * OBSP_), TPB, 0, stream>>>(x, xb, B_, OBS_, OBSP_);
  cvt_transpose_batched<<<nb((long)H_ * OBSP_), TPB, 0, stream>>>(enc_W1, encW1t, 1, OBS_, H_, OBSP_);
  cvt_transpose_batched<<<nb((long)H_ * H_), TPB, 0, stream>>>(enc_W2, encW2t, 1, H_, H_, H_);
  cvt_transpose_batched<<<nb((long)L1_ * 64 * H_), TPB, 0, stream>>>(a_rW, a_rWt, L1_, H_, 64, H_);
  cvt_transpose_batched<<<nb((long)L1_ * 64 * H_), TPB, 0, stream>>>(c_rW, c_rWt, L1_, H_, 64, H_);
  cvt_transpose_batched<<<nb((long)L1_ * M_ * H_ * H_), TPB, 0, stream>>>(a_W, a_Wt, L1_ * M_, H_, H_, H_);
  cvt_transpose_batched<<<nb((long)L1_ * M_ * H_ * H_), TPB, 0, stream>>>(c_W, c_Wt, L1_ * M_, H_, H_, H_);

  // encoder: f_s = tanh(tanh(x@W1+b1)@W2+b2)
  dim3 ge(B_ / 128, H_ / 64, 1);
  gemm_bf16_wmma<<<ge, TPB, 0, stream>>>(xb, OBSP_, 0, encW1t, OBSP_, 0, enc_b1, 0,
                                         h1, nullptr, H_, 0, OBSP_, 2);
  gemm_bf16_wmma<<<ge, TPB, 0, stream>>>(h1, H_, 0, encW2t, H_, 0, enc_b2, 0,
                                         fs, nullptr, H_, 0, H_, 2);
  context_relu<<<nb((long)B_ * H_), TPB, 0, stream>>>(fs, task_idx, task_emb, relc);

  // branches: 0 = critic, 1 = actor
  for (int br = 0; br < 2; br++) {
    const unsigned short* rWt = (br == 0) ? c_rWt : a_rWt;
    const float*          rb  = (br == 0) ? c_rb  : a_rb;
    const unsigned short* Wm  = (br == 0) ? c_Wt  : a_Wt;
    const float*          bm  = (br == 0) ? c_b   : a_b;

    // routing: logits -> softmax(probs over groups of 8)
    for (int l = 0; l < L1_; l++) {
      dim3 gr(B_ / 128, 1, 1);
      gemm_bf16_wmma<<<gr, TPB, 0, stream>>>(relc, H_, 0, rWt + (long)l * 64 * H_, H_, 0,
                                             rb + (long)l * 64, 0,
                                             nullptr, probs + (long)l * B_ * 64, 64, 0, H_, 0);
      softmax8_inplace<<<nb((long)B_ * M_), TPB, 0, stream>>>(probs + (long)l * B_ * 64, B_ * M_);
    }

    // module layers with soft combine
    for (int l = 0; l < L1_; l++) {
      dim3 gm(B_ / 128, H_ / 64, M_);
      const unsigned short* Ain = (l == 0) ? fs : G;
      int  lda = (l == 0) ? H_ : M_ * H_;
      long sAm = (l == 0) ? 0 : H_;
      gemm_bf16_wmma<<<gm, TPB, 0, stream>>>(Ain, lda, sAm,
                                             Wm + (long)l * M_ * H_ * H_, H_, (long)H_ * H_,
                                             bm + (long)l * M_ * H_, H_,
                                             O, nullptr, M_ * H_, H_, H_, 1);
      combine_kernel<<<B_, TPB, 0, stream>>>(probs + (long)l * B_ * 64, O, G);
    }

    if (br == 0) critic_final<<<B_, TPB, 0, stream>>>(G, c_Wf, c_bf, out);
    else         actor_final<<<B_, TPB, 0, stream>>>(G, a_Wf, a_bf, logstd, out);
  }
}
